// ConfigurableSpikingNeuralNet_43628277793225
// MI455X (gfx1250) — compile-verified, matrix operand order checked by reading, not run on hardware
//
#include <hip/hip_runtime.h>
#include <stdint.h>

// ---------------------------------------------------------------------------
// Types for CDNA5 WMMA
// ---------------------------------------------------------------------------
typedef __attribute__((ext_vector_type(16))) __bf16 bf16x16;
typedef __attribute__((ext_vector_type(8)))  float  f32x8;

union FragBF16 {
    bf16x16 v;
    uint4   q[2];   // 32 bytes = 2 x b128 loads
};

// Problem constants
constexpr int T_   = 50;
constexpr int B_   = 256;
constexpr int IN_  = 700;
constexpr int H_   = 1024;
constexpr int OUT_ = 20;
constexpr int INP_  = 704;  // layer-0 K padded to /32
constexpr int OUTP_ = 32;   // layer-3 N padded to /16

#define BETA      0.9f
#define THRESH    1.0f
#define BF16_ONE  ((unsigned short)0x3F80)

__device__ __forceinline__ unsigned short f32_to_bf16(float f) {
    unsigned int u = __float_as_uint(f);
    unsigned int rounding = 0x7FFFu + ((u >> 16) & 1u);
    return (unsigned short)((u + rounding) >> 16);
}

__device__ __forceinline__ void lif_update(float mp, float cur, float& mn, float& spk) {
    float reset = (mp - THRESH > 0.0f) ? 1.0f : 0.0f;
    mn  = BETA * mp + cur - reset * THRESH;
    spk = (mn - THRESH > 0.0f) ? 1.0f : 0.0f;
}

// ---------------------------------------------------------------------------
// Monotonic grid barrier (all 74 WGs resident: far below MI455X WGP count).
// Per-wave store/load counters + per-CU caches mean EVERY thread must fence
// (device scope) around the arrive/wait, not just the signalling thread.
// ---------------------------------------------------------------------------
__device__ __forceinline__ void grid_sync(unsigned int* cnt, unsigned int target) {
    __threadfence();                 // release: publish this wave's stores
    __syncthreads();
    if (threadIdx.x == 0) {
        __hip_atomic_fetch_add(cnt, 1u, __ATOMIC_RELEASE, __HIP_MEMORY_SCOPE_AGENT);
        while (__hip_atomic_load(cnt, __ATOMIC_RELAXED, __HIP_MEMORY_SCOPE_AGENT) < target)
            __builtin_amdgcn_s_sleep(1);
    }
    __syncthreads();
    __threadfence();                 // acquire: invalidate stale cached lines
}

// ---------------------------------------------------------------------------
// Conversions
// ---------------------------------------------------------------------------
__global__ void cvt_pad_bf16(const float* __restrict__ src,
                             unsigned short* __restrict__ dst,
                             int srows, int scols, int dcols, long total) {
    long i = (long)blockIdx.x * blockDim.x + threadIdx.x;
    if (i >= total) return;
    int c = (int)(i % dcols);
    int r = (int)(i / dcols);
    float v = (r < srows && c < scols) ? src[(long)r * scols + c] : 0.0f;
    dst[i] = f32_to_bf16(v);
}

__global__ void pad_bias_f32(const float* __restrict__ src, float* __restrict__ dst,
                             int n, int dn) {
    int i = blockIdx.x * blockDim.x + threadIdx.x;
    if (i < dn) dst[i] = (i < n) ? src[i] : 0.0f;
}

// ---------------------------------------------------------------------------
// WMMA GEMM tile:  D[tile] = A[M,K] * W[N,K]^T (+bias), optional fused LIF
//   A, W row-major / K-contiguous, K % 32 == 0, rows 16B-aligned.
//   ISA 16-bit frag layout: lane&15 = row, lane>>4 = K-half, 2 x b128 per frag.
// ---------------------------------------------------------------------------
template<int WM, int WN, int WAVES_M, int WAVES_N, bool FUSE>
__device__ __forceinline__ void gemm_tile(
        const unsigned short* __restrict__ A, int lda,
        const unsigned short* __restrict__ W, int ldw,
        const float* __restrict__ bias, int K,
        int tileMbase, int tileNbase,
        float* __restrict__ C, int ldc,
        float* __restrict__ mem_state,
        float* __restrict__ spk_rec,
        float* __restrict__ mem_rec,
        unsigned short* __restrict__ spk_b,
        int nOut) {
    const int wave  = threadIdx.x >> 5;
    const int lane  = threadIdx.x & 31;
    const int waveM = wave / WAVES_N;
    const int waveN = wave % WAVES_N;
    const int tileM = tileMbase + waveM * (WM * 16);
    const int tileN = tileNbase + waveN * (WN * 16);

    const int row = lane & 15;
    const int kh  = lane >> 4;

    f32x8 acc[WM][WN];
#pragma unroll
    for (int i = 0; i < WM; ++i)
#pragma unroll
        for (int j = 0; j < WN; ++j) {
            f32x8 z = {0.f, 0.f, 0.f, 0.f, 0.f, 0.f, 0.f, 0.f};
            acc[i][j] = z;
        }

    const unsigned short* aBase[WM];
#pragma unroll
    for (int i = 0; i < WM; ++i)
        aBase[i] = A + (size_t)(tileM + i * 16 + row) * lda + kh * 8;
    const unsigned short* bBase[WN];
#pragma unroll
    for (int j = 0; j < WN; ++j)
        bBase[j] = W + (size_t)(tileN + j * 16 + row) * ldw + kh * 8;

    for (int k0 = 0; k0 < K; k0 += 32) {
        FragBF16 af[WM], bf[WN];
#pragma unroll
        for (int i = 0; i < WM; ++i) {
            af[i].q[0] = *(const uint4*)(aBase[i] + k0);
            af[i].q[1] = *(const uint4*)(aBase[i] + k0 + 16);
        }
#pragma unroll
        for (int j = 0; j < WN; ++j) {
            bf[j].q[0] = *(const uint4*)(bBase[j] + k0);
            bf[j].q[1] = *(const uint4*)(bBase[j] + k0 + 16);
        }
#pragma unroll
        for (int i = 0; i < WM; ++i)
#pragma unroll
            for (int j = 0; j < WN; ++j)
                acc[i][j] = __builtin_amdgcn_wmma_f32_16x16x32_bf16(
                    false, af[i].v, false, bf[j].v,
                    (short)0, acc[i][j], false, false);
    }

    // Epilogue. D layout: n = lane&15, m = r + 8*(lane>>4)
    const int nloc  = lane & 15;
    const int mhalf = (lane >> 4) << 3;
#pragma unroll
    for (int i = 0; i < WM; ++i) {
#pragma unroll
        for (int j = 0; j < WN; ++j) {
            const int n  = tileN + j * 16 + nloc;
            const int mb = tileM + i * 16 + mhalf;
            const float bv = bias[n];
            if (FUSE) {
                if (n < nOut) {
#pragma unroll
                    for (int r = 0; r < 8; ++r) {
                        const size_t idx = (size_t)(mb + r) * nOut + n;
                        float curv = acc[i][j][r] + bv;
                        float mn, spk;
                        lif_update(mem_state[idx], curv, mn, spk);
                        mem_state[idx] = mn;
                        mem_rec[idx]   = mn;
                        spk_rec[idx]   = spk;
                        if (spk_b)
                            spk_b[idx] = (spk > 0.0f) ? BF16_ONE : (unsigned short)0;
                    }
                }
            } else {
#pragma unroll
                for (int r = 0; r < 8; ++r)
                    C[(size_t)(mb + r) * ldc + n] = acc[i][j][r] + bv;
            }
        }
    }
}

// ---------------------------------------------------------------------------
// Layer-0 currents for ALL timesteps (time-parallel, machine-wide)
// ---------------------------------------------------------------------------
__global__ __launch_bounds__(256)
void gemm0_kernel(const unsigned short* __restrict__ A, int lda,
                  const unsigned short* __restrict__ W, int ldw,
                  const float* __restrict__ bias, int K,
                  float* __restrict__ C, int ldc) {
    gemm_tile<4, 2, 2, 4, false>(A, lda, W, ldw, bias, K,
                                 blockIdx.y * 128, blockIdx.x * 128,
                                 C, ldc, nullptr, nullptr, nullptr, nullptr, 0);
}

// ---------------------------------------------------------------------------
// Persistent pipelined SNN kernel.
//   Phase s runs concurrently:  LIF0@t=s | GEMM1+LIF@t=s-1 | GEMM2+LIF@t=s-2 |
//   GEMM3+LIF@t=s-3, double-buffered spike buffers indexed by t&1.
//   WGs: [0,8) LIF0, [8,40) GEMM1, [40,72) GEMM2, [72,74) GEMM3.
// ---------------------------------------------------------------------------
__global__ __launch_bounds__(256)
void snn_pipeline(const float* __restrict__ cur0,
                  const unsigned short* __restrict__ w1b,
                  const unsigned short* __restrict__ w2b,
                  const unsigned short* __restrict__ w3p,
                  const float* __restrict__ b1,
                  const float* __restrict__ b2,
                  const float* __restrict__ b3p,
                  unsigned short* __restrict__ spk0b,   // [2][B*H]
                  unsigned short* __restrict__ spk1b,   // [2][B*H]
                  unsigned short* __restrict__ spk2b,   // [2][B*H]
                  float* __restrict__ mem0, float* __restrict__ mem1,
                  float* __restrict__ mem2, float* __restrict__ mem3,
                  float* __restrict__ out,
                  unsigned int* __restrict__ cnt) {
    const int wg = blockIdx.x;
    const size_t TBH = (size_t)T_ * B_ * H_;
    const size_t TBO = (size_t)T_ * B_ * OUT_;
    float* spk0r = out;
    float* spk1r = out + TBH;
    float* spk2r = out + 2 * TBH;
    float* spk3r = out + 3 * TBH;
    float* mem0r = out + 3 * TBH + TBO;
    float* mem1r = out + 4 * TBH + TBO;
    float* mem2r = out + 5 * TBH + TBO;
    float* mem3r = out + 6 * TBH + TBO;

    constexpr int NWG    = 74;
    constexpr int PHASES = T_ + 3;
    unsigned int target = 0;

    for (int s = 0; s < PHASES; ++s) {
        if (wg < 8) {
            // ---- LIF layer 0, t = s (current precomputed by gemm0_kernel)
            const int t = s;
            if (t < T_) {
                const float4*  cur = (const float4*)(cur0 + (size_t)t * B_ * H_);
                float4*        mst = (float4*)mem0;
                float4*        sr  = (float4*)(spk0r + (size_t)t * B_ * H_);
                float4*        mr  = (float4*)(mem0r + (size_t)t * B_ * H_);
                ushort4*       sb  = (ushort4*)(spk0b + (size_t)(t & 1) * B_ * H_);
                const int tid = wg * 256 + threadIdx.x;         // 0..2047
                constexpr int N4 = (B_ * H_) / 4;
                for (int i = tid; i < N4; i += 8 * 256) {
                    float4 c = cur[i];
                    float4 m = mst[i];
                    float4 mn, sp;
                    lif_update(m.x, c.x, mn.x, sp.x);
                    lif_update(m.y, c.y, mn.y, sp.y);
                    lif_update(m.z, c.z, mn.z, sp.z);
                    lif_update(m.w, c.w, mn.w, sp.w);
                    mst[i] = mn;
                    mr[i]  = mn;
                    sr[i]  = sp;
                    ushort4 sv;
                    sv.x = (sp.x > 0.f) ? BF16_ONE : (unsigned short)0;
                    sv.y = (sp.y > 0.f) ? BF16_ONE : (unsigned short)0;
                    sv.z = (sp.z > 0.f) ? BF16_ONE : (unsigned short)0;
                    sv.w = (sp.w > 0.f) ? BF16_ONE : (unsigned short)0;
                    sb[i] = sv;
                }
            }
        } else if (wg < 40) {
            // ---- GEMM1 + LIF, t = s-1   (32 WGs, 64x128 tiles)
            const int t = s - 1;
            if (t >= 0 && t < T_) {
                const int g = wg - 8;
                gemm_tile<2, 2, 2, 4, true>(
                    spk0b + (size_t)(t & 1) * B_ * H_, H_, w1b, H_, b1, H_,
                    (g >> 3) * 64, (g & 7) * 128, nullptr, 0,
                    mem1, spk1r + (size_t)t * B_ * H_, mem1r + (size_t)t * B_ * H_,
                    spk1b + (size_t)(t & 1) * B_ * H_, H_);
            }
        } else if (wg < 72) {
            // ---- GEMM2 + LIF, t = s-2   (32 WGs, 64x128 tiles)
            const int t = s - 2;
            if (t >= 0 && t < T_) {
                const int g = wg - 40;
                gemm_tile<2, 2, 2, 4, true>(
                    spk1b + (size_t)(t & 1) * B_ * H_, H_, w2b, H_, b2, H_,
                    (g >> 3) * 64, (g & 7) * 128, nullptr, 0,
                    mem2, spk2r + (size_t)t * B_ * H_, mem2r + (size_t)t * B_ * H_,
                    spk2b + (size_t)(t & 1) * B_ * H_, H_);
            }
        } else {
            // ---- GEMM3 + LIF, t = s-3   (2 WGs, 128x32 tiles, N padded)
            const int t = s - 3;
            if (t >= 0 && t < T_) {
                const int g = wg - 72;
                gemm_tile<1, 2, 8, 1, true>(
                    spk2b + (size_t)(t & 1) * B_ * H_, H_, w3p, H_, b3p, H_,
                    g * 128, 0, nullptr, 0,
                    mem3, spk3r + (size_t)t * B_ * OUT_, mem3r + (size_t)t * B_ * OUT_,
                    nullptr, OUT_);
            }
        }
        target += NWG;
        if (s != PHASES - 1) grid_sync(cnt, target);
    }
}

// ---------------------------------------------------------------------------
// Host-side orchestration (9 launches total)
// ---------------------------------------------------------------------------
extern "C" void kernel_launch(void* const* d_in, const int* in_sizes, int n_in,
                              void* d_out, int out_size, void* d_ws, size_t ws_size,
                              hipStream_t stream) {
    const float* x  = (const float*)d_in[0];
    const float* W0 = (const float*)d_in[1];
    const float* b0 = (const float*)d_in[2];
    const float* W1 = (const float*)d_in[3];
    const float* b1 = (const float*)d_in[4];
    const float* W2 = (const float*)d_in[5];
    const float* b2 = (const float*)d_in[6];
    const float* W3 = (const float*)d_in[7];
    const float* b3 = (const float*)d_in[8];
    float* out = (float*)d_out;
    (void)in_sizes; (void)n_in; (void)out_size; (void)ws_size;

    char* ws = (char*)d_ws;
    size_t off = 0;
    auto alloc = [&](size_t bytes) -> char* {
        char* p = ws + off;
        off += (bytes + 255) & ~(size_t)255;
        return p;
    };
    unsigned short* xp    = (unsigned short*)alloc((size_t)T_ * B_ * INP_ * 2);
    unsigned short* w0p   = (unsigned short*)alloc((size_t)H_ * INP_ * 2);
    unsigned short* w1b   = (unsigned short*)alloc((size_t)H_ * H_ * 2);
    unsigned short* w2b   = (unsigned short*)alloc((size_t)H_ * H_ * 2);
    unsigned short* w3p   = (unsigned short*)alloc((size_t)OUTP_ * H_ * 2);
    float*          b3p   = (float*)alloc(OUTP_ * 4);
    float*          cur0  = (float*)alloc((size_t)T_ * B_ * H_ * 4);
    unsigned short* spk0b = (unsigned short*)alloc((size_t)2 * B_ * H_ * 2);
    unsigned short* spk1b = (unsigned short*)alloc((size_t)2 * B_ * H_ * 2);
    unsigned short* spk2b = (unsigned short*)alloc((size_t)2 * B_ * H_ * 2);
    // membrane states + sync counter, zeroed by one async memset
    size_t stateBytes = ((size_t)3 * B_ * H_ + (size_t)B_ * OUT_) * 4;
    float* mem0 = (float*)alloc(stateBytes + 256);
    float* mem1 = mem0 + (size_t)B_ * H_;
    float* mem2 = mem1 + (size_t)B_ * H_;
    float* mem3 = mem2 + (size_t)B_ * H_;
    unsigned int* cnt = (unsigned int*)((char*)mem0 + stateBytes);

    hipMemsetAsync(mem0, 0, stateBytes + sizeof(unsigned int), stream);

    const int thr = 256;
    auto nblk = [&](long n) { return (int)((n + thr - 1) / thr); };

    cvt_pad_bf16<<<nblk((long)T_ * B_ * INP_), thr, 0, stream>>>(x, xp, T_ * B_, IN_, INP_, (long)T_ * B_ * INP_);
    cvt_pad_bf16<<<nblk((long)H_ * INP_), thr, 0, stream>>>(W0, w0p, H_, IN_, INP_, (long)H_ * INP_);
    cvt_pad_bf16<<<nblk((long)H_ * H_), thr, 0, stream>>>(W1, w1b, H_, H_, H_, (long)H_ * H_);
    cvt_pad_bf16<<<nblk((long)H_ * H_), thr, 0, stream>>>(W2, w2b, H_, H_, H_, (long)H_ * H_);
    cvt_pad_bf16<<<nblk((long)OUTP_ * H_), thr, 0, stream>>>(W3, w3p, OUT_, H_, H_, (long)OUTP_ * H_);
    pad_bias_f32<<<1, OUTP_, 0, stream>>>(b3, b3p, OUT_, OUTP_);

    // Time-parallel layer-0 GEMM: cur0[T*B, H] = xp * w0p^T + b0
    {
        dim3 grid(H_ / 128, (T_ * B_) / 128);
        gemm0_kernel<<<grid, 256, 0, stream>>>(xp, INP_, w0p, INP_, b0, INP_, cur0, H_);
    }

    // Single persistent kernel runs the whole 50-step recurrence, pipelined.
    snn_pipeline<<<74, 256, 0, stream>>>(cur0, w1b, w2b, w3p, b1, b2, b3p,
                                         spk0b, spk1b, spk2b,
                                         mem0, mem1, mem2, mem3,
                                         out, cnt);
}
